// TreeLSTM_8976481649002
// MI455X (gfx1250) — matrix operand order repaired and checked by reading, not compiled, not run on hardware
//
#include <hip/hip_runtime.h>

// ---------------------------------------------------------------------------
// MI455X (gfx1250) TreeLSTM: all GEMMs via v_wmma_f32_16x16x32_bf16 (wave32).
// f32 inputs are downconverted to bf16 in LDS (TF32-style), f32 accumulation.
// B/A fragments loaded as ds_load_b128 pairs; column tiles processed in pairs
// (two independent accumulators) to overlap LDS traffic with WMMA issue.
// ---------------------------------------------------------------------------

typedef __attribute__((ext_vector_type(16))) __bf16    v16bf;
typedef __attribute__((ext_vector_type(8)))  float     v8f;
typedef __attribute__((ext_vector_type(4)))  unsigned  v4u;

union FragU { v16bf v; v4u q[2]; unsigned u[8]; };

__device__ inline v8f v8f_zero() {
  v8f r;
#pragma unroll
  for (int i = 0; i < 8; ++i) r[i] = 0.0f;
  return r;
}

__device__ inline v16bf fragA_zero() {
  FragU f;
#pragma unroll
  for (int i = 0; i < 8; ++i) f.u[i] = 0u;
  return f.v;
}

// A fragment (16x32 bf16): lane&15 = row; lane half selects K offset 0/8;
// VGPR 0-3 -> K {kb..kb+7}, VGPR 4-7 -> K {kb+16..kb+23}. (ISA 7.12.2)
// Both 16-byte chunks are 16B-aligned (stride bytes % 16 == 0, 2*kb % 16 == 0).
__device__ inline v16bf ldsA(const __bf16* base, int stride, int row0, int k0, int lane) {
  int r  = row0 + (lane & 15);
  int kb = k0 + ((lane & 16) ? 8 : 0);
  FragU f;
  f.q[0] = *(const v4u*)(base + r * stride + kb);
  f.q[1] = *(const v4u*)(base + r * stride + kb + 16);
  return f.v;
}

// B fragment (32x16 bf16): lane&15 = col; lane half selects K offset 0/8;
// VGPR 0-3 -> K {kb..kb+7}, VGPR 4-7 -> K {kb+16..kb+23}. LDS holds B transposed: bT[col*stride + k].
__device__ inline v16bf ldsB(const __bf16* base, int stride, int col0, int k0, int lane) {
  int c  = col0 + (lane & 15);
  int kb = k0 + ((lane & 16) ? 8 : 0);
  FragU f;
  f.q[0] = *(const v4u*)(base + c * stride + kb);
  f.q[1] = *(const v4u*)(base + c * stride + kb + 16);
  return f.v;
}

__device__ inline v8f wmma_bf16(v16bf a, v16bf b, v8f c) {
  return __builtin_amdgcn_wmma_f32_16x16x32_bf16(false, a, false, b, (short)0, c,
                                                 false, false);
}

// Two independent column tiles per pass: breaks the single-accumulator WMMA
// dependency chain and lets ds_load_b128 for tile B overlap tile A's WMMAs.
template <int KSTEPS>
__device__ inline void gemm_pair(const v16bf (&afr)[KSTEPS], const __bf16* B,
                                 int bstr, int colA, int colB, int lane,
                                 v8f& accA, v8f& accB) {
#pragma unroll
  for (int kk = 0; kk < KSTEPS; ++kk) {
    v16bf bA = ldsB(B, bstr, colA, kk * 32, lane);
    v16bf bB = ldsB(B, bstr, colB, kk * 32, lane);
    accA = wmma_bf16(afr[kk], bA, accA);
    accB = wmma_bf16(afr[kk], bB, accB);
  }
}

__device__ inline float sigf(float x) { return 1.0f / (1.0f + __expf(-x)); }

#define H128 128
#define STR  136   // 128 + 8 bf16 pad (row = 272 B, 16B-aligned)
#define STRW 264   // 256 + 8 bf16 pad (row = 528 B, 16B-aligned)

// ===========================================================================
// K1: x_tild = x @ Wiou (N x 384), n_tild = x @ Qf (N x 128). 128 nodes/block.
// ===========================================================================
__global__ __launch_bounds__(256)
void k1_proj_x(const float* __restrict__ x, const float* __restrict__ Wiou,
               const float* __restrict__ Qf, float* __restrict__ xt,
               float* __restrict__ nt, int N) {
  extern __shared__ char smem[];
  __bf16* xA = (__bf16*)smem;                       // 128 x STR
  __bf16* wB = (__bf16*)(smem + H128 * STR * 2);    // 128 x STR (transposed)
  const int tid = threadIdx.x, lane = tid & 31, wv = tid >> 5;
  const int nb0 = blockIdx.x * 128;
  const int r0 = wv * 16, hi8 = (lane & 16) ? 8 : 0, ln = lane & 15;

  for (int i = tid; i < 128 * 128; i += 256) {
    int r = i >> 7, c = i & 127;
    int g = nb0 + r; if (g >= N) g = N - 1;
    xA[r * STR + c] = (__bf16)x[(long)g * H128 + c];
  }
  __syncthreads();

  v16bf afr[4];
#pragma unroll
  for (int kk = 0; kk < 4; ++kk) afr[kk] = ldsA(xA, STR, r0, kk * 32, lane);

  for (int chunk = 0; chunk < 4; ++chunk) {
    for (int i = tid; i < 128 * 128; i += 256) {
      int k = i >> 7, n = i & 127;
      float wv_ = (chunk < 3) ? Wiou[(long)k * 384 + chunk * 128 + n]
                              : Qf[(long)k * 128 + n];
      wB[n * STR + k] = (__bf16)wv_;
    }
    __syncthreads();

    float* dst = (chunk < 3) ? xt : nt;
    const long ldd = (chunk < 3) ? 384 : 128;
    const int coff = (chunk < 3) ? chunk * 128 : 0;

#pragma unroll
    for (int cp = 0; cp < 4; ++cp) {
      v8f a0 = v8f_zero(), a1 = v8f_zero();
      gemm_pair<4>(afr, wB, STR, cp * 32, cp * 32 + 16, lane, a0, a1);
      int col0 = cp * 32 + ln, col1 = cp * 32 + 16 + ln;
#pragma unroll
      for (int j = 0; j < 8; ++j) {
        int g = nb0 + r0 + hi8 + j;
        if (g < N) {
          dst[(long)g * ldd + coff + col0] = a0[j];
          dst[(long)g * ldd + coff + col1] = a1[j];
        }
      }
    }
    __syncthreads();   // GEMM reads of wB done before next chunk overwrites
  }
}

// ===========================================================================
// K2: per 16 nodes (128 mail rows): hk = hm@Kf; logits/softmax; v = hk@vf with
// fused avec0/avec1 reduction; masked hm@Of + hm@Zf -> sigmoid -> c_red.
// ===========================================================================
__global__ __launch_bounds__(256)
void k2_mail(const float* __restrict__ h_mail, const float* __restrict__ c_mail,
             const float* __restrict__ Kf, const float* __restrict__ vf,
             const float* __restrict__ Of, const float* __restrict__ Zf,
             const float* __restrict__ Vf, const int* __restrict__ etype,
             const float* __restrict__ nt, float* __restrict__ av0,
             float* __restrict__ av1, float* __restrict__ cred, int N) {
  extern __shared__ char smem[];
  __bf16* hm   = (__bf16*)smem;                         // 128 x STR  (h_mail)
  __bf16* hkb  = (__bf16*)(smem + 1 * H128 * STR * 2);  // 128 x STR  (hk)
  __bf16* wB   = (__bf16*)(smem + 2 * H128 * STR * 2);  // 128 x STR  (weight^T)
  float*  ntl  = (float*)(smem + 3 * H128 * STR * 2);   // 16 x 128
  float*  vfl  = ntl + 16 * 128;                        // 128
  float*  logit= vfl + 128;                             // 128
  float*  coef0= logit + 128;                           // 128
  float*  coef1= coef0 + 128;                           // 128

  const int tid = threadIdx.x, lane = tid & 31, wv = tid >> 5;
  const int m0 = blockIdx.x * 16;
  const long rbase = (long)m0 * 8;
  const int r0 = wv * 16, hi8 = (lane & 16) ? 8 : 0, ln = lane & 15;
  const int nodeL = m0 + ((r0 + hi8) >> 3);   // node this lane reduces into

  // prefetch the c_mail tile (consumed only in the final epilogue):
  // 64 KB, one global_prefetch_b8 per 256B chunk per thread
  {
    const float* cm0 = c_mail + rbase * H128;
    for (int i = tid * 64; i < 128 * 128; i += 256 * 64)
      __builtin_prefetch(cm0 + i, 0, 3);
  }

  // stage h_mail tile (f32 -> bf16), n_tild rows, Vf, Kf^T
  for (int i = tid; i < 128 * 128; i += 256) {
    int r = i >> 7, c = i & 127;
    hm[r * STR + c] = (__bf16)h_mail[(rbase + r) * H128 + c];
  }
  for (int i = tid; i < 16 * 128; i += 256) ntl[i] = nt[(long)m0 * H128 + i];
  if (tid < 128) vfl[tid] = Vf[tid];
  for (int i = tid; i < 128 * 128; i += 256) {
    int k = i >> 7, n = i & 127;
    wB[n * STR + k] = (__bf16)Kf[i];
  }
  __syncthreads();

  // h_mail A fragments: reused by GEMM1 (Kf) and masked GEMMs (Of/Zf)
  v16bf ahm[4];
#pragma unroll
  for (int kk = 0; kk < 4; ++kk) ahm[kk] = ldsA(hm, STR, r0, kk * 32, lane);

  // GEMM1: hk = hm @ Kf  -> hkb (bf16)
#pragma unroll
  for (int cp = 0; cp < 4; ++cp) {
    v8f a0 = v8f_zero(), a1 = v8f_zero();
    gemm_pair<4>(ahm, wB, STR, cp * 32, cp * 32 + 16, lane, a0, a1);
    int col0 = cp * 32 + ln, col1 = cp * 32 + 16 + ln;
#pragma unroll
    for (int j = 0; j < 8; ++j) {
      hkb[(r0 + hi8 + j) * STR + col0] = (__bf16)a0[j];
      hkb[(r0 + hi8 + j) * STR + col1] = (__bf16)a1[j];
    }
  }
  __syncthreads();

  // logits: tanh((n_tild + hk) . Vf) per mail row
  if (tid < 128) {
    int r = tid, m = r >> 3;
    float s = 0.0f;
    for (int c = 0; c < 128; ++c)
      s += (ntl[m * 128 + c] + (float)hkb[r * STR + c]) * vfl[c];
    logit[r] = tanhf(s);
  }
  __syncthreads();

  // softmax over k per node + fused e-masking coefficients
  if (tid < 16) {
    int m = tid;
    float mx = -1e30f;
    for (int k = 0; k < 8; ++k) mx = fmaxf(mx, logit[m * 8 + k]);
    float ex[8], den = 0.0f;
    for (int k = 0; k < 8; ++k) { ex[k] = __expf(logit[m * 8 + k] - mx); den += ex[k]; }
    float inv = 1.0f / den;
    for (int k = 0; k < 8; ++k) {
      float w = ex[k] * inv;
      float e = (float)etype[(long)(m0 + m) * 8 + k];
      coef1[m * 8 + k] = w * e;
      coef0[m * 8 + k] = w * (1.0f - e);
    }
  }
  for (int i = tid; i < 128 * 128; i += 256) {      // vf^T
    int k = i >> 7, n = i & 127;
    wB[n * STR + k] = (__bf16)vf[i];
  }
  __syncthreads();

  float c0v[8], c1v[8];
#pragma unroll
  for (int j = 0; j < 8; ++j) { c0v[j] = coef0[r0 + hi8 + j]; c1v[j] = coef1[r0 + hi8 + j]; }

  // GEMM2: v = hk @ vf, epilogue reduces over k with softmax*e weights
  {
    v16bf ahk[4];
#pragma unroll
    for (int kk = 0; kk < 4; ++kk) ahk[kk] = ldsA(hkb, STR, r0, kk * 32, lane);
#pragma unroll
    for (int cp = 0; cp < 4; ++cp) {
      v8f a0 = v8f_zero(), a1 = v8f_zero();
      gemm_pair<4>(ahk, wB, STR, cp * 32, cp * 32 + 16, lane, a0, a1);
      float s00 = 0.f, s01 = 0.f, s10 = 0.f, s11 = 0.f;
#pragma unroll
      for (int j = 0; j < 8; ++j) {
        s00 += c0v[j] * a0[j]; s10 += c1v[j] * a0[j];
        s01 += c0v[j] * a1[j]; s11 += c1v[j] * a1[j];
      }
      int col0 = cp * 32 + ln, col1 = cp * 32 + 16 + ln;
      av0[(long)nodeL * H128 + col0] = s00;
      av1[(long)nodeL * H128 + col0] = s10;
      av0[(long)nodeL * H128 + col1] = s01;
      av1[(long)nodeL * H128 + col1] = s11;
    }
  }
  __syncthreads();

  // f-gate: (hm.e)@Of + (hm.(1-e))@Zf, row-masked A fragments (e in {0,1})
  const int emask = etype[rbase + r0 + (lane & 15)];  // A-frag row fixed per lane
  v16bf aM[4];
  v8f accf[8];
#pragma unroll
  for (int ct = 0; ct < 8; ++ct) accf[ct] = v8f_zero();

  for (int i = tid; i < 128 * 128; i += 256) {      // Of^T
    int k = i >> 7, n = i & 127;
    wB[n * STR + k] = (__bf16)Of[i];
  }
  __syncthreads();
#pragma unroll
  for (int kk = 0; kk < 4; ++kk) aM[kk] = emask ? ahm[kk] : fragA_zero();
#pragma unroll
  for (int cp = 0; cp < 4; ++cp)
    gemm_pair<4>(aM, wB, STR, cp * 32, cp * 32 + 16, lane,
                 accf[2 * cp], accf[2 * cp + 1]);
  __syncthreads();

  for (int i = tid; i < 128 * 128; i += 256) {      // Zf^T
    int k = i >> 7, n = i & 127;
    wB[n * STR + k] = (__bf16)Zf[i];
  }
  __syncthreads();
#pragma unroll
  for (int kk = 0; kk < 4; ++kk) aM[kk] = emask ? fragA_zero() : ahm[kk];
#pragma unroll
  for (int cp = 0; cp < 4; ++cp)
    gemm_pair<4>(aM, wB, STR, cp * 32, cp * 32 + 16, lane,
                 accf[2 * cp], accf[2 * cp + 1]);

  // epilogue: f = sigmoid(acc); c_red = sum_k f * c_mail
#pragma unroll
  for (int ct = 0; ct < 8; ++ct) {
    int col = (ct >> 1) * 32 + (ct & 1) * 16 + ln;
    float s = 0.0f;
#pragma unroll
    for (int j = 0; j < 8; ++j) {
      long grow = rbase + r0 + hi8 + j;
      s += sigf(accf[ct][j]) * c_mail[grow * H128 + col];
    }
    cred[(long)nodeL * H128 + col] = s;
  }
}

// ===========================================================================
// K3: h_til0/1 = modify-scaled avec@Wd/Wa -> h_tild (LDS bf16, 128x256);
// iou = h_tild@Uiou + x_tild + biou done per i/o/u column-triple with fused
// LSTM gate epilogue writing h and c. 128 nodes/block.
// ===========================================================================
__global__ __launch_bounds__(256)
void k3_out(const float* __restrict__ av0, const float* __restrict__ av1,
            const float* __restrict__ Wa, const float* __restrict__ Wd,
            const float* __restrict__ Uiou, const float* __restrict__ biou,
            const float* __restrict__ xt, const float* __restrict__ cred,
            const int* __restrict__ etype, float* __restrict__ out, int N) {
  extern __shared__ char smem[];
  __bf16* htld = (__bf16*)smem;                                   // 128 x STRW
  __bf16* aT   = (__bf16*)(smem + H128 * STRW * 2);               // 128 x STR
  __bf16* wB   = (__bf16*)(smem + H128 * STRW * 2 + H128 * STR * 2);
  float*  mdf  = (float*)(smem + H128 * STRW * 2 + 2 * H128 * STR * 2); // 128

  const int tid = threadIdx.x, lane = tid & 31, wv = tid >> 5;
  const int nb0 = blockIdx.x * 128;
  const int r0 = wv * 16, hi8 = (lane & 16) ? 8 : 0, ln = lane & 15;

  if (tid < 128) {
    int g = nb0 + tid; if (g >= N) g = N - 1;
    int s = 0;
    for (int k = 0; k < 8; ++k) s += etype[(long)g * 8 + k];
    mdf[tid] = (float)s * 0.125f;
  }

  // pass 0: h_til0 = modify * (avec0 @ Wd) -> cols [0,128)
  // pass 1: h_til1 = (1-modify) * (avec1 @ Wa) -> cols [128,256)
  for (int pass = 0; pass < 2; ++pass) {
    const float* av = pass ? av1 : av0;
    const float* W  = pass ? Wa  : Wd;
    __syncthreads();
    for (int i = tid; i < 128 * 128; i += 256) {
      int r = i >> 7, c = i & 127;
      int g = nb0 + r; if (g >= N) g = N - 1;
      aT[r * STR + c] = (__bf16)av[(long)g * H128 + c];
    }
    for (int i = tid; i < 128 * 128; i += 256) {
      int k = i >> 7, n = i & 127;
      wB[n * STR + k] = (__bf16)W[i];
    }
    __syncthreads();

    v16bf afr[4];
#pragma unroll
    for (int kk = 0; kk < 4; ++kk) afr[kk] = ldsA(aT, STR, r0, kk * 32, lane);

#pragma unroll
    for (int cp = 0; cp < 4; ++cp) {
      v8f a0 = v8f_zero(), a1 = v8f_zero();
      gemm_pair<4>(afr, wB, STR, cp * 32, cp * 32 + 16, lane, a0, a1);
      int col0 = cp * 32 + ln, col1 = cp * 32 + 16 + ln;
#pragma unroll
      for (int j = 0; j < 8; ++j) {
        int r = r0 + hi8 + j;
        float m = mdf[r];
        float scale = pass ? (1.0f - m) : m;
        htld[r * STRW + pass * 128 + col0] = (__bf16)(scale * a0[j]);
        htld[r * STRW + pass * 128 + col1] = (__bf16)(scale * a1[j]);
      }
    }
  }

  // gate GEMM: K=256, i/o/u column triples per 16-col tile
  for (int j8 = 0; j8 < 8; ++j8) {
    __syncthreads();
    for (int i = tid; i < 48 * 256; i += 256) {     // Uiou^T triple chunk
      int cl = i >> 8, k = i & 255;
      int g = cl >> 4, t = cl & 15;
      wB[cl * STRW + k] = (__bf16)Uiou[(long)k * 384 + g * 128 + j8 * 16 + t];
    }
    __syncthreads();
    v8f ai = v8f_zero(), ao = v8f_zero(), au = v8f_zero();
#pragma unroll
    for (int k0 = 0; k0 < 256; k0 += 32) {
      v16bf a = ldsA(htld, STRW, r0, k0, lane);
      ai = wmma_bf16(a, ldsB(wB, STRW,  0, k0, lane), ai);
      ao = wmma_bf16(a, ldsB(wB, STRW, 16, k0, lane), ao);
      au = wmma_bf16(a, ldsB(wB, STRW, 32, k0, lane), au);
    }
    int colh = j8 * 16 + ln;
    float bi = biou[colh], bo = biou[128 + colh], bu = biou[256 + colh];
#pragma unroll
    for (int j = 0; j < 8; ++j) {
      long g = nb0 + r0 + hi8 + j;
      if (g < N) {
        float ig = sigf(xt[g * 384 + colh]       + ai[j] + bi);
        float og = sigf(xt[g * 384 + 128 + colh] + ao[j] + bo);
        float ug = tanhf(xt[g * 384 + 256 + colh] + au[j] + bu);
        float cc = ig * ug + cred[g * H128 + colh];
        float hh = og * tanhf(cc);
        out[g * H128 + colh] = hh;                      // h
        out[(long)N * H128 + g * H128 + colh] = cc;     // c
      }
    }
  }
}

// ===========================================================================
extern "C" void kernel_launch(void* const* d_in, const int* in_sizes, int n_in,
                              void* d_out, int out_size, void* d_ws, size_t ws_size,
                              hipStream_t stream) {
  const float* x      = (const float*)d_in[0];
  const float* h_mail = (const float*)d_in[1];
  const float* c_mail = (const float*)d_in[2];
  const float* Wiou   = (const float*)d_in[3];
  const float* Uiou   = (const float*)d_in[4];
  const float* biou   = (const float*)d_in[5];
  const float* Of     = (const float*)d_in[6];
  const float* Zf     = (const float*)d_in[7];
  const float* Qf     = (const float*)d_in[8];
  const float* Kf     = (const float*)d_in[9];
  const float* vf     = (const float*)d_in[10];
  const float* Wa     = (const float*)d_in[11];
  const float* Wd     = (const float*)d_in[12];
  const float* Vf     = (const float*)d_in[13];
  const int*   etype  = (const int*)d_in[14];
  float* out = (float*)d_out;

  const int N = in_sizes[0] / 128;   // 50000

  // workspace layout (f32): n_tild | x_tild | avec0 | avec1 | c_red
  float* nt   = (float*)d_ws;
  float* xt   = nt  + (size_t)N * 128;
  float* av0  = xt  + (size_t)N * 384;
  float* av1  = av0 + (size_t)N * 128;
  float* cred = av1 + (size_t)N * 128;

  const size_t s1 = (size_t)2 * H128 * STR * 2;                                 // ~70 KB
  const size_t s2 = (size_t)3 * H128 * STR * 2 + 16 * 128 * 4 + 4 * 128 * 4;    // ~112 KB
  const size_t s3 = (size_t)H128 * STRW * 2 + 2 * (size_t)H128 * STR * 2 + 512; // ~134 KB

  (void)hipFuncSetAttribute((const void*)k1_proj_x,
      hipFuncAttributeMaxDynamicSharedMemorySize, (int)s1);
  (void)hipFuncSetAttribute((const void*)k2_mail,
      hipFuncAttributeMaxDynamicSharedMemorySize, (int)s2);
  (void)hipFuncSetAttribute((const void*)k3_out,
      hipFuncAttributeMaxDynamicSharedMemorySize, (int)s3);

  const int gbNodes128 = (N + 127) / 128;
  const int gbNodes16  = (N + 15) / 16;

  k1_proj_x<<<gbNodes128, 256, s1, stream>>>(x, Wiou, Qf, xt, nt, N);
  k2_mail<<<gbNodes16, 256, s2, stream>>>(h_mail, c_mail, Kf, vf, Of, Zf, Vf,
                                          etype, nt, av0, av1, cred, N);
  k3_out<<<gbNodes128, 256, s3, stream>>>(av0, av1, Wa, Wd, Uiou, biou, xt,
                                          cred, etype, out, N);
}
